// Critic_21998822490677
// MI455X (gfx1250) — compile-verified, matrix operand order checked by reading, not run on hardware
//
#include <hip/hip_runtime.h>

// ---------------------------------------------------------------------------
// Problem constants (from the reference)
// ---------------------------------------------------------------------------
#define N_AGENTS 8
#define OBS 128
#define ACT 16
#define HEADS 4
#define EMB 256
#define ATT 64
#define B1 512
#define B2 32
#define SAMP (B1 * B2)            // 16384 samples per agent
#define ROWS (N_AGENTS * SAMP)    // 131072 total rows
#define KOA 160                   // OBS+ACT=144 padded to multiple of 32
#define NQKV (3 * HEADS * ATT)    // 768 = [q(256) | k(256) | v(256)]
#define FIN (EMB + HEADS * ATT)   // 512 = [xi(256) | si(256)]

typedef __attribute__((ext_vector_type(16))) _Float16 v16h;
typedef __attribute__((ext_vector_type(8)))  float    v8f;
typedef unsigned long long u64;

union HFrag { v16h v; unsigned u[8]; };
union FFrag { v8f  v; float    f[8]; };

// CDNA5 async global->LDS copy, 16B per lane, tracked by ASYNCcnt.
// ldsAddr = low 32 bits of the generic LDS pointer (LDS aperture truncates
// to addr[31:0], ISA 10.2); gaddr = per-lane 64-bit global address.
__device__ __forceinline__ void async_copy16(unsigned ldsAddr, const void* gaddr) {
  asm volatile("global_load_async_to_lds_b128 %0, %1, off"
               :: "v"(ldsAddr), "v"((u64)gaddr) : "memory");
}

// ---------------------------------------------------------------------------
// Conversion kernels: build f16 activations (padded) and f16 transposed weights
// ---------------------------------------------------------------------------
__global__ __launch_bounds__(256) void pack_oa_kernel(
    const float* __restrict__ obs, const float* __restrict__ act,
    _Float16* __restrict__ oa) {
  size_t idx = (size_t)blockIdx.x * 256 + threadIdx.x;
  size_t total = (size_t)ROWS * KOA;
  if (idx >= total) return;
  size_t row = idx / KOA;
  int c = (int)(idx % KOA);
  float v = 0.f;
  if (c < OBS)            v = obs[row * OBS + c];
  else if (c < OBS + ACT) v = act[row * ACT + (c - OBS)];
  oa[idx] = (_Float16)v;
}

// gWt[n][o][i] (i padded to 160) = gW[n][i][o]
__global__ __launch_bounds__(256) void conv_gWt_kernel(
    const float* __restrict__ gW, _Float16* __restrict__ gWt) {
  size_t idx = (size_t)blockIdx.x * 256 + threadIdx.x;
  size_t total = (size_t)N_AGENTS * EMB * KOA;
  if (idx >= total) return;
  int n = (int)(idx / (EMB * KOA));
  int r = (int)(idx % (EMB * KOA));
  int o = r / KOA, i = r % KOA;
  float v = (i < OBS + ACT) ? gW[((size_t)n * (OBS + ACT) + i) * EMB + o] : 0.f;
  gWt[idx] = (_Float16)v;
}

// sWt[n][o][i] = sW[n][i][o]
__global__ __launch_bounds__(256) void conv_sWt_kernel(
    const float* __restrict__ sW, _Float16* __restrict__ sWt) {
  size_t idx = (size_t)blockIdx.x * 256 + threadIdx.x;
  size_t total = (size_t)N_AGENTS * EMB * OBS;
  if (idx >= total) return;
  int n = (int)(idx / (EMB * OBS));
  int r = (int)(idx % (EMB * OBS));
  int o = r / OBS, i = r % OBS;
  sWt[idx] = (_Float16)sW[((size_t)n * OBS + i) * EMB + o];
}

// qkvWt[p*256 + h*64 + a][e] = W{q,k,v}[h][e][a]
__global__ __launch_bounds__(256) void conv_qkvWt_kernel(
    const float* __restrict__ Wq, const float* __restrict__ Wk,
    const float* __restrict__ Wv, _Float16* __restrict__ Wt) {
  size_t idx = (size_t)blockIdx.x * 256 + threadIdx.x;
  size_t total = (size_t)NQKV * EMB;
  if (idx >= total) return;
  int n = (int)(idx / EMB);
  int e = (int)(idx % EMB);
  int p = n >> 8;          // 0:q 1:k 2:v
  int h = (n & 255) >> 6;
  int a = n & 63;
  const float* W = (p == 0) ? Wq : (p == 1) ? Wk : Wv;
  Wt[idx] = (_Float16)W[((size_t)h * EMB + e) * ATT + a];
}

// f1Wt[n][o][i] = f1W[n][i][o]
__global__ __launch_bounds__(256) void conv_f1Wt_kernel(
    const float* __restrict__ f1W, _Float16* __restrict__ f1Wt) {
  size_t idx = (size_t)blockIdx.x * 256 + threadIdx.x;
  size_t total = (size_t)N_AGENTS * EMB * FIN;
  if (idx >= total) return;
  int n = (int)(idx / (EMB * FIN));
  int r = (int)(idx % (EMB * FIN));
  int o = r / FIN, i = r % FIN;
  f1Wt[idx] = (_Float16)f1W[((size_t)n * FIN + i) * EMB + o];
}

// ---------------------------------------------------------------------------
// Unified WMMA GEMM:  out = act( A[M,K] @ Wt[N,K]^T + bias )
//   128x64 output tile / block, 128 threads = 4 waves; each wave computes a
//   32x64 sub-tile (2 A fragments x 4 B fragments = 8 wmma per K-step, each
//   B fragment reused by two wmmas).
//   Double-buffered LDS, filled by GLOBAL_LOAD_ASYNC_TO_LDS_B128 (ASYNCcnt):
//   prefetch K-chunk i+1 while v_wmma_f32_16x16x32_f16 consumes chunk i.
//   Each wave issues exactly 6 async b128s per chunk (4 A + 2 B) ->
//   s_wait_asynccnt 0x6 retires the older group before the consume barrier.
// ---------------------------------------------------------------------------
__global__ __launch_bounds__(128) void wmma_gemm_kernel(
    const _Float16* __restrict__ A,  long long strideA,  int lda,
    const _Float16* __restrict__ Wt, long long strideW,
    const float* __restrict__ bias,  int strideBias,
    _Float16* __restrict__ outH,     long long strideOut, int ldc,
    int M, int N, int K, int lreluStart) {
  __shared__ __align__(16) _Float16 As[2][128 * 32];
  __shared__ __align__(16) _Float16 Bs[2][64 * 32];

  const int z = blockIdx.z;
  const int m0 = blockIdx.y * 128;
  const int n0 = blockIdx.x * 64;
  A    += (size_t)z * strideA;
  Wt   += (size_t)z * strideW;
  outH += (size_t)z * strideOut;
  if (bias) bias += (size_t)z * strideBias;

  const int t = threadIdx.x;           // 0..127
  const int wave = t >> 5;             // 0..3
  const int lane = t & 31;
  const int lrow = lane & 15;
  const int khalf = (lane >> 4) * 8;   // 0 or 8

  // --- async-copy addressing: rows are 32 halfs = 2 x 16B chunks per... no:
  //     each 32-half row = 4 x 16B chunks.  A tile = 128 rows -> 512 chunks,
  //     B tile = 64 rows -> 256 chunks.  Thread t owns A chunks t+{0,128,256,384}
  //     (rows cr+{0,32,64,96}) and B chunks t+{0,128} (rows cr+{0,32}).
  const int cr = t >> 2;               // 0..31
  const int ch = (t & 3) * 8;          // half offset within the 32-half row
  const _Float16* gA0 = A + (size_t)(m0 + cr) * lda + ch;
  const _Float16* gA1 = A + (size_t)(m0 + cr + 32) * lda + ch;
  const _Float16* gA2 = A + (size_t)(m0 + cr + 64) * lda + ch;
  const _Float16* gA3 = A + (size_t)(m0 + cr + 96) * lda + ch;
  const _Float16* gB0 = Wt + (size_t)(n0 + cr) * K + ch;
  const _Float16* gB1 = Wt + (size_t)(n0 + cr + 32) * K + ch;
  unsigned lA[2], lB[2];
  lA[0] = (unsigned)(u64)(const void*)&As[0][0] + t * 16;
  lA[1] = (unsigned)(u64)(const void*)&As[1][0] + t * 16;
  lB[0] = (unsigned)(u64)(const void*)&Bs[0][0] + t * 16;
  lB[1] = (unsigned)(u64)(const void*)&Bs[1][0] + t * 16;

  FFrag acc0[4], acc1[4];
#pragma unroll
  for (int c = 0; c < 4; c++) {
    acc0[c].v = (v8f){0.f, 0.f, 0.f, 0.f, 0.f, 0.f, 0.f, 0.f};
    acc1[c].v = (v8f){0.f, 0.f, 0.f, 0.f, 0.f, 0.f, 0.f, 0.f};
  }

  const int nsteps = K >> 5;

  // prologue: prefetch chunk 0 into buffer 0 (6 async b128 per wave)
  async_copy16(lA[0],        gA0);
  async_copy16(lA[0] + 2048, gA1);
  async_copy16(lA[0] + 4096, gA2);
  async_copy16(lA[0] + 6144, gA3);
  async_copy16(lB[0],        gB0);
  async_copy16(lB[0] + 2048, gB1);

  for (int i = 0; i < nsteps; i++) {
    const int cur = i & 1;
    if (i + 1 < nsteps) {
      const int k1 = (i + 1) << 5;
      async_copy16(lA[cur ^ 1],        gA0 + k1);
      async_copy16(lA[cur ^ 1] + 2048, gA1 + k1);
      async_copy16(lA[cur ^ 1] + 4096, gA2 + k1);
      async_copy16(lA[cur ^ 1] + 6144, gA3 + k1);
      async_copy16(lB[cur ^ 1],        gB0 + k1);
      async_copy16(lB[cur ^ 1] + 2048, gB1 + k1);
      asm volatile("s_wait_asynccnt 0x6" ::: "memory");  // retire older group
    } else {
      asm volatile("s_wait_asynccnt 0x0" ::: "memory");
    }
    __syncthreads();  // whole tile visible to all waves

    // A fragments: lane<16 holds K {0..7,16..23}, lane>=16 holds K {8..15,24..31}
    HFrag a0, a1;
    const unsigned* Au0 = (const unsigned*)(&As[cur][(wave * 32 + lrow) * 32]);
    const unsigned* Au1 = (const unsigned*)(&As[cur][(wave * 32 + 16 + lrow) * 32]);
#pragma unroll
    for (int v = 0; v < 8; v++) {
      int kk = ((v & 4) ? 16 : 0) + khalf + (v & 3) * 2;
      a0.u[v] = Au0[kk >> 1];
      a1.u[v] = Au1[kk >> 1];
    }
#pragma unroll
    for (int c = 0; c < 4; c++) {
      HFrag b;
      const unsigned* Bu = (const unsigned*)(&Bs[cur][(c * 16 + lrow) * 32]);
#pragma unroll
      for (int v = 0; v < 8; v++) {
        int kk = ((v & 4) ? 16 : 0) + khalf + (v & 3) * 2;
        b.u[v] = Bu[kk >> 1];
      }
      acc0[c].v = __builtin_amdgcn_wmma_f32_16x16x32_f16(
          false, a0.v, false, b.v, (short)0, acc0[c].v, false, false);
      acc1[c].v = __builtin_amdgcn_wmma_f32_16x16x32_f16(
          false, a1.v, false, b.v, (short)0, acc1[c].v, false, false);
    }
    __syncthreads();  // all reads done before this buffer is refilled
  }

  // epilogue: bias + ranged leaky-ReLU, store f16
#pragma unroll
  for (int c = 0; c < 4; c++) {
    int ncol = n0 + c * 16 + lrow;
    float bv = bias ? bias[ncol] : 0.f;
    bool doRelu = (ncol >= lreluStart);
#pragma unroll
    for (int i = 0; i < 8; i++) {
      int r0 = m0 + wave * 32 + i + (lane >> 4) * 8;
      float x = acc0[c].f[i] + bv;
      if (doRelu && x < 0.f) x *= 0.01f;
      outH[(size_t)r0 * ldc + ncol] = (_Float16)x;
      int r1 = r0 + 16;
      float y = acc1[c].f[i] + bv;
      if (doRelu && y < 0.f) y *= 0.01f;
      outH[(size_t)r1 * ldc + ncol] = (_Float16)y;
    }
  }
}

// ---------------------------------------------------------------------------
// Exclude-self attention over agents.  One block per sample (x,y).
// qkv row (agent n): [q(h,a) | k(h,a) | v(h,a)] with col = p*256 + h*64 + a.
// 8-thread groups per (h,n): thread j computes logit vs agent m=j, then its
// own softmax weight, then 8 of the 64 output features.
// ---------------------------------------------------------------------------
__global__ __launch_bounds__(256) void attn_kernel(
    const _Float16* __restrict__ qkv, _Float16* __restrict__ f_in) {
  __shared__ float sh[3 * N_AGENTS * HEADS * ATT];  // q | k | v, [p][n][h*64+a]
  __shared__ float lg[256];
  __shared__ float al[256];
  const int s = blockIdx.x;

  for (int i = threadIdx.x; i < N_AGENTS * NQKV; i += 256) {
    int n = i / NQKV, col = i % NQKV;
    int p = col >> 8, ha = col & 255;
    sh[p * 2048 + n * 256 + ha] =
        (float)qkv[((size_t)n * SAMP + s) * NQKV + col];
  }
  __syncthreads();

  const int g = threadIdx.x >> 3;  // 0..31 : (h,n)
  const int j = threadIdx.x & 7;   // agent m
  const int h = g >> 3, n = g & 7;

  const float* q = sh + n * 256 + h * 64;
  const float* k = sh + 2048 + j * 256 + h * 64;
  float dot = 0.f;
#pragma unroll
  for (int a = 0; a < ATT; a++) dot += q[a] * k[a];
  lg[threadIdx.x] = dot * 0.17677669529663687f;  // 1/sqrt(B2=32)
  __syncthreads();

  float mx = -3.4e38f;
#pragma unroll
  for (int m = 0; m < 8; m++) if (m != n) { float v = lg[g * 8 + m]; mx = v > mx ? v : mx; }
  float sum = 0.f;
#pragma unroll
  for (int m = 0; m < 8; m++) if (m != n) sum += expf(lg[g * 8 + m] - mx);
  al[threadIdx.x] = (j == n) ? 0.f : expf(lg[threadIdx.x] - mx) / sum;
  __syncthreads();

#pragma unroll
  for (int aa = 0; aa < 8; aa++) {
    int a = j * 8 + aa;
    float x = 0.f;
#pragma unroll
    for (int m = 0; m < 8; m++)
      x += al[g * 8 + m] * sh[4096 + m * 256 + h * 64 + a];
    f_in[((size_t)n * SAMP + s) * FIN + h * 64 + a] = (_Float16)x;
  }
}

// ---------------------------------------------------------------------------
// Final stage: id = argmax(act row); out = h1 . f2W[n][:, id] + f2b[n][id]
// One wave per row (32 lanes over K=256), shuffle reduction.
// ---------------------------------------------------------------------------
__global__ __launch_bounds__(256) void final_dot_kernel(
    const float* __restrict__ act, const _Float16* __restrict__ h1,
    const float* __restrict__ f2W, const float* __restrict__ f2b,
    float* __restrict__ out) {
  const int row = blockIdx.x * 8 + (threadIdx.x >> 5);
  const int lane = threadIdx.x & 31;
  const int n = row >> 14;  // /SAMP

  const float* arow = act + (size_t)row * ACT;
  int id = 0; float best = arow[0];
#pragma unroll
  for (int i = 1; i < ACT; i++) { float v = arow[i]; if (v > best) { best = v; id = i; } }

  const _Float16* hrow = h1 + (size_t)row * EMB;
  const float* w = f2W + (size_t)n * EMB * ACT + id;
  float accv = 0.f;
#pragma unroll
  for (int i = 0; i < 8; i++) {
    int kk = lane + 32 * i;
    accv += (float)hrow[kk] * w[(size_t)kk * ACT];
  }
#pragma unroll
  for (int off = 16; off > 0; off >>= 1) accv += __shfl_xor(accv, off, 32);
  if (lane == 0) out[row] = accv + f2b[n * ACT + id];
}

// ---------------------------------------------------------------------------
// Host launcher
// ---------------------------------------------------------------------------
extern "C" void kernel_launch(void* const* d_in, const int* in_sizes, int n_in,
                              void* d_out, int out_size, void* d_ws, size_t ws_size,
                              hipStream_t stream) {
  const float* obs = (const float*)d_in[0];
  const float* act = (const float*)d_in[1];
  const float* gW  = (const float*)d_in[2];
  const float* gb  = (const float*)d_in[3];
  const float* sW  = (const float*)d_in[4];
  const float* sb  = (const float*)d_in[5];
  const float* Wq  = (const float*)d_in[6];
  const float* Wk  = (const float*)d_in[7];
  const float* Wv  = (const float*)d_in[8];
  const float* f1W = (const float*)d_in[9];
  const float* f1b = (const float*)d_in[10];
  const float* f2W = (const float*)d_in[11];
  const float* f2b = (const float*)d_in[12];
  float* out = (float*)d_out;

  // workspace layout (256B aligned)
  size_t off = 0;
  auto alloc = [&](size_t bytes) { size_t p = off; off += (bytes + 255) & ~(size_t)255; return p; };
  char* ws = (char*)d_ws;
  _Float16* oaH   = (_Float16*)(ws + alloc((size_t)ROWS * KOA * 2));   // 40 MB
  _Float16* eiH   = (_Float16*)(ws + alloc((size_t)ROWS * EMB * 2));   // 64 MB
  _Float16* qkvH  = (_Float16*)(ws + alloc((size_t)ROWS * NQKV * 2));  // 192 MB
  _Float16* finH  = (_Float16*)(ws + alloc((size_t)ROWS * FIN * 2));   // 128 MB
  _Float16* gWtH  = (_Float16*)(ws + alloc((size_t)N_AGENTS * EMB * KOA * 2));
  _Float16* sWtH  = (_Float16*)(ws + alloc((size_t)N_AGENTS * EMB * OBS * 2));
  _Float16* qWtH  = (_Float16*)(ws + alloc((size_t)NQKV * EMB * 2));
  _Float16* f1WtH = (_Float16*)(ws + alloc((size_t)N_AGENTS * EMB * FIN * 2));
  _Float16* h1H   = qkvH;  // qkv dead after attn; f1 runs after attn on same stream

  // 1) conversions
  {
    size_t tot = (size_t)ROWS * KOA;
    pack_oa_kernel<<<dim3((tot + 255) / 256), 256, 0, stream>>>(obs, act, oaH);
  }
  {
    size_t tot = (size_t)N_AGENTS * EMB * KOA;
    conv_gWt_kernel<<<dim3((tot + 255) / 256), 256, 0, stream>>>(gW, gWtH);
  }
  {
    size_t tot = (size_t)N_AGENTS * EMB * OBS;
    conv_sWt_kernel<<<dim3((tot + 255) / 256), 256, 0, stream>>>(sW, sWtH);
  }
  {
    size_t tot = (size_t)NQKV * EMB;
    conv_qkvWt_kernel<<<dim3((tot + 255) / 256), 256, 0, stream>>>(Wq, Wk, Wv, qWtH);
  }
  {
    size_t tot = (size_t)N_AGENTS * EMB * FIN;
    conv_f1Wt_kernel<<<dim3((tot + 255) / 256), 256, 0, stream>>>(f1W, f1WtH);
  }

  // 2) ei = lrelu(oa @ gW + gb)      [per-agent, K padded to 160]
  wmma_gemm_kernel<<<dim3(EMB / 64, SAMP / 128, N_AGENTS), 128, 0, stream>>>(
      oaH, (long long)SAMP * KOA, KOA,
      gWtH, (long long)EMB * KOA,
      gb, EMB,
      eiH, (long long)SAMP * EMB, EMB,
      SAMP, EMB, KOA, /*lrelu all*/ 0);

  // 3) qkv = ei @ [Wq|Wk|Wv]         [one big GEMM; lrelu only on v cols >=512]
  wmma_gemm_kernel<<<dim3(NQKV / 64, ROWS / 128, 1), 128, 0, stream>>>(
      eiH, 0, EMB,
      qWtH, 0,
      nullptr, 0,
      qkvH, 0, NQKV,
      ROWS, NQKV, EMB, /*lrelu v only*/ 2 * EMB);

  // 4) si = lrelu(obs @ sW + sb)  -> written straight into f_in[:, 256:512]
  wmma_gemm_kernel<<<dim3(EMB / 64, SAMP / 128, N_AGENTS), 128, 0, stream>>>(
      oaH, (long long)SAMP * KOA, KOA,   // obs is the first 128 cols of oa
      sWtH, (long long)EMB * OBS,
      sb, EMB,
      finH + EMB, (long long)SAMP * FIN, FIN,
      SAMP, EMB, OBS, 0);

  // 5) attention -> f_in[:, 0:256]
  attn_kernel<<<dim3(SAMP), 256, 0, stream>>>(qkvH, finH);

  // 6) h1 = lrelu(f_in @ f1W + f1b)   (h1 reuses qkv scratch)
  wmma_gemm_kernel<<<dim3(EMB / 64, SAMP / 128, N_AGENTS), 128, 0, stream>>>(
      finH, (long long)SAMP * FIN, FIN,
      f1WtH, (long long)EMB * FIN,
      f1b, EMB,
      h1H, (long long)SAMP * EMB, EMB,
      SAMP, EMB, FIN, 0);

  // 7) final: argmax-select + single-column f2 dot
  final_dot_kernel<<<dim3(ROWS / 8), 256, 0, stream>>>(act, h1H, f2W, f2b, out);

  (void)in_sizes; (void)n_in; (void)out_size; (void)ws_size;
}
